// GCNN2_81063212744721
// MI455X (gfx1250) — matrix operand order repaired and checked by reading, not compiled, not run on hardware
//
#include <hip/hip_runtime.h>
#include <hip/hip_bf16.h>

#define NNODES 10000
#define NEDGES 160000
#define FEAT   512
#define OUTF   128
#define NGRAPH 64

typedef __attribute__((ext_vector_type(16))) __bf16 v16bf;
typedef __attribute__((ext_vector_type(8)))  float  v8f;

union BFrag {
    unsigned short u[16];
    uint4          q[2];
    v16bf          v;
};

static __device__ inline unsigned short f2bf(float f) {
    unsigned u = __builtin_bit_cast(unsigned, f);
    u += 0x7FFFu + ((u >> 16) & 1u);           // round-to-nearest-even
    return (unsigned short)(u >> 16);
}

// ---------------------------------------------------------------- utility
__global__ void fill_f32(float* __restrict__ p, float v, int n) {
    int i = blockIdx.x * blockDim.x + threadIdx.x;
    if (i < n) p[i] = v;
}

__global__ void cvt_bf16(const float* __restrict__ in, unsigned short* __restrict__ out, int n) {
    int i = blockIdx.x * blockDim.x + threadIdx.x;
    if (i < n) out[i] = f2bf(in[i]);
}

__global__ void deg_accum(const int* __restrict__ dst, float* __restrict__ deg, int e) {
    int i = blockIdx.x * blockDim.x + threadIdx.x;
    if (i < e) unsafeAtomicAdd(&deg[dst[i]], 1.0f);
}

__global__ void deg_rsqrt(float* __restrict__ deg, int n) {
    int i = blockIdx.x * blockDim.x + threadIdx.x;
    if (i < n) deg[i] = rsqrtf(deg[i]);        // deg >= 1 (self loop)
}

// Pack W (K x NC, f32 row-major) into bf16 WMMA B-fragment order:
// P[((kt*NTILES + ntile)*32 + lane)*16 + e] =
//     bf16(W[(kt*32 + 8*(lane>>4) + (e&7) + 16*(e>>3))*NC + ntile*16 + (lane&15)])
template <int NC>
__global__ void pack_w(const float* __restrict__ W, unsigned short* __restrict__ P, int K) {
    const int NTILES = NC / 16;
    int tid = blockIdx.x * blockDim.x + threadIdx.x;
    if (tid >= K * NC) return;
    int e     = tid & 15;
    int lane  = (tid >> 4) & 31;
    int rest  = tid >> 9;
    int ntile = rest % NTILES;
    int kt    = rest / NTILES;
    int half  = lane >> 4;
    int col   = ntile * 16 + (lane & 15);
    int k     = kt * 32 + half * 8 + (e & 7) + 16 * (e >> 3);
    P[tid] = f2bf(W[(size_t)k * NC + col]);
}

// ------------------------------------------------------- WMMA bf16 GEMM
// C[M x NC] = A[M x K] * Bpacked[K x NC] (+bias).
// A is bf16 row-major; Bp is bf16 in fragment order. M%16==0, K%32==0.
// Block: 256 threads = 8 waves; block b covers rows [16b,16b+16);
// wave w covers columns [w*16*NT, (w+1)*16*NT). Requires NC == 128*NT.
template <int NC, int NT>
__global__ __launch_bounds__(256) void gemm_wmma_bf16(
    const unsigned short* __restrict__ A, const unsigned short* __restrict__ Bp,
    float* __restrict__ C, int K, const float* __restrict__ bias)
{
    const int NTILES = NC / 16;
    const int lane = threadIdx.x & 31;
    const int wave = threadIdx.x >> 5;
    const int half = lane >> 4;
    const int l15  = lane & 15;
    const int m0   = blockIdx.x * 16;
    const int nt0  = wave * NT;

    v8f acc[NT] = {};

    // Loop-carried pointers: strides are compile-time constants so B-tile
    // offsets (t*1024B) and the second half of each fragment (+16B) fold
    // into the 24-bit instruction offset.
    const unsigned short* ap = A + (size_t)(m0 + l15) * K + half * 8;
    const unsigned short* bp = Bp + ((size_t)nt0 * 32 + lane) * 16;

    for (int kt = 0; kt < K / 32; ++kt) {
        // A fragment (16-bit A 16x32 layout):
        // elems 0..7  <-> K = kt*32 + 8*half + 0..7   (contiguous bf16 -> b128)
        // elems 8..15 <-> K = kt*32 + 16 + 8*half + 0..7
        BFrag a;
        a.q[0] = *(const uint4*)(ap);
        a.q[1] = *(const uint4*)(ap + 16);

        BFrag b[NT];
        #pragma unroll
        for (int t = 0; t < NT; ++t) {
            b[t].q[0] = *(const uint4*)(bp + t * 512);
            b[t].q[1] = *(const uint4*)(bp + t * 512 + 8);
        }

        #pragma unroll
        for (int t = 0; t < NT; ++t) {
            acc[t] = __builtin_amdgcn_wmma_f32_16x16x32_bf16(
                false, a.v, false, b[t].v, (short)0, acc[t], false, false);
        }

        ap += 32;              // next K-step of A
        bp += NTILES * 512;    // next K-step of packed B
    }

    // store: 32-bit C/D layout: VGPR r -> M = 8*half + r, N = l15.
    #pragma unroll
    for (int t = 0; t < NT; ++t) {
        int col = (nt0 + t) * 16 + l15;
        float badd = bias ? bias[col] : 0.0f;
        #pragma unroll
        for (int r = 0; r < 8; ++r) {
            int row = m0 + half * 8 + r;
            C[(size_t)row * NC + col] = acc[t][r] + badd;
        }
    }
}

// ------------------------------------------------------- GCN aggregation
// Self-loop contribution: hact[i,f] = hlin[i,f] * dinv[i]^2
__global__ void self_init(const float* __restrict__ hlin, const float* __restrict__ dinv,
                          float* __restrict__ hact) {
    int tid = blockIdx.x * blockDim.x + threadIdx.x;   // N*512 threads
    int i = tid >> 9;
    float di = dinv[i];
    hact[tid] = hlin[tid] * di * di;
}

// Edge scatter: one block per edge, 256 threads x 2 feats.
__global__ __launch_bounds__(256) void edge_scatter(
    const float* __restrict__ hlin, float* __restrict__ hact,
    const float* __restrict__ dinv,
    const int* __restrict__ src, const int* __restrict__ dst)
{
    int e = blockIdx.x;
    int s = src[e];
    int d = dst[e];
    float w = dinv[s] * dinv[d];
    int f = threadIdx.x * 2;
    float2 hv = *(const float2*)(hlin + (size_t)s * FEAT + f);
    float* out = hact + (size_t)d * FEAT + f;
    unsafeAtomicAdd(out,     hv.x * w);
    unsafeAtomicAdd(out + 1, hv.y * w);
}

// Layer-1 epilogue: bias + relu, emit bf16 (input of GEMM-2).
__global__ void bias_relu_bf16(const float* __restrict__ h, const float* __restrict__ b,
                               unsigned short* __restrict__ out) {
    int tid = blockIdx.x * blockDim.x + threadIdx.x;   // N*512 threads
    float v = h[tid] + b[tid & (FEAT - 1)];
    out[tid] = f2bf(v > 0.0f ? v : 0.0f);
}

// Layer-2 epilogue: bias + relu in f32 (pooling consumes f32).
__global__ void bias_relu(float* __restrict__ h, const float* __restrict__ b) {
    int tid = blockIdx.x * blockDim.x + threadIdx.x;   // N*512 threads
    float v = h[tid] + b[tid & (FEAT - 1)];
    h[tid] = v > 0.0f ? v : 0.0f;
}

__global__ void pool_scatter(const float* __restrict__ h, const int* __restrict__ batch,
                             float* __restrict__ pooled) {
    int tid = blockIdx.x * blockDim.x + threadIdx.x;   // N*512 threads
    int i = tid >> 9;
    int g = batch[i];
    unsafeAtomicAdd(&pooled[(size_t)g * FEAT + (tid & (FEAT - 1))], h[tid]);
}

// ---------------------------------------------------------------- launch
extern "C" void kernel_launch(void* const* d_in, const int* in_sizes, int n_in,
                              void* d_out, int out_size, void* d_ws, size_t ws_size,
                              hipStream_t stream) {
    const float* x    = (const float*)d_in[0];   // [N,512]
    const float* W1   = (const float*)d_in[1];   // [512,512]
    const float* b1   = (const float*)d_in[2];   // [512]
    const float* W2   = (const float*)d_in[3];   // [512,512]
    const float* b2   = (const float*)d_in[4];   // [512]
    const float* Wl   = (const float*)d_in[5];   // [512,128]
    const float* bl   = (const float*)d_in[6];   // [128]
    const int*   eidx = (const int*)d_in[7];     // [2,E] row-major
    const int*   bat  = (const int*)d_in[8];     // [N]
    const int* srcI = eidx;
    const int* dstI = eidx + NEDGES;

    float* outF    = (float*)d_out;              // pooled [64,512] then out [64,128]
    float* pooled  = outF;
    float* outFin  = outF + NGRAPH * FEAT;

    // workspace layout
    char* ws = (char*)d_ws;
    float* dinv = (float*)ws;                                        // N floats (64KB slot)
    float* bufA = (float*)(ws + 65536);                              // N*512 f32
    float* bufB = bufA + (size_t)NNODES * FEAT;                      // N*512 f32
    unsigned short* pW1 = (unsigned short*)(bufB + (size_t)NNODES * FEAT);
    unsigned short* pW2 = pW1 + FEAT * FEAT;
    unsigned short* pWl = pW2 + FEAT * FEAT;
    unsigned short* abf = pWl + FEAT * OUTF;                         // N*512 bf16 (GEMM A input)
    unsigned short* pbf = abf + (size_t)NNODES * FEAT;               // 64*512 bf16 (pooled)

    const int NF = NNODES * FEAT;                                    // 5,120,000

    // 1) degrees -> dinv
    fill_f32<<<(NNODES + 255) / 256, 256, 0, stream>>>(dinv, 1.0f, NNODES);
    deg_accum<<<(NEDGES + 255) / 256, 256, 0, stream>>>(dstI, dinv, NEDGES);
    deg_rsqrt<<<(NNODES + 255) / 256, 256, 0, stream>>>(dinv, NNODES);

    // 2) pack weights to bf16 fragment order
    pack_w<FEAT><<<(FEAT * FEAT + 255) / 256, 256, 0, stream>>>(W1, pW1, FEAT);
    pack_w<FEAT><<<(FEAT * FEAT + 255) / 256, 256, 0, stream>>>(W2, pW2, FEAT);
    pack_w<OUTF><<<(FEAT * OUTF + 255) / 256, 256, 0, stream>>>(Wl, pWl, FEAT);

    // 3) layer 1: hlin1 = x @ W1 ; agg ; +b1 relu -> bf16
    cvt_bf16<<<NF / 256, 256, 0, stream>>>(x, abf, NF);
    gemm_wmma_bf16<FEAT, 4><<<NNODES / 16, 256, 0, stream>>>(abf, pW1, bufA, FEAT, nullptr);
    self_init<<<NF / 256, 256, 0, stream>>>(bufA, dinv, bufB);
    edge_scatter<<<NEDGES, 256, 0, stream>>>(bufA, bufB, dinv, srcI, dstI);
    bias_relu_bf16<<<NF / 256, 256, 0, stream>>>(bufB, b1, abf);

    // 4) layer 2: hlin2 = h1 @ W2 ; agg ; +b2 relu (f32)
    gemm_wmma_bf16<FEAT, 4><<<NNODES / 16, 256, 0, stream>>>(abf, pW2, bufA, FEAT, nullptr);
    self_init<<<NF / 256, 256, 0, stream>>>(bufA, dinv, bufB);
    edge_scatter<<<NEDGES, 256, 0, stream>>>(bufA, bufB, dinv, srcI, dstI);
    bias_relu<<<NF / 256, 256, 0, stream>>>(bufB, b2);

    // 5) pool into d_out[0 : 64*512]
    fill_f32<<<(out_size + 255) / 256, 256, 0, stream>>>(outF, 0.0f, out_size);
    pool_scatter<<<NF / 256, 256, 0, stream>>>(bufB, bat, pooled);

    // 6) out = pooled @ Wlin + blin into d_out[64*512 : ]
    cvt_bf16<<<(NGRAPH * FEAT + 255) / 256, 256, 0, stream>>>(pooled, pbf, NGRAPH * FEAT);
    gemm_wmma_bf16<OUTF, 1><<<NGRAPH / 16, 256, 0, stream>>>(pbf, pWl, outFin, FEAT, bl);
}
